// Curvphormer_4346506903650
// MI455X (gfx1250) — matrix operand order repaired and checked by reading, not compiled, not run on hardware
//
#include <hip/hip_runtime.h>
#include <hip/hip_bf16.h>

typedef __attribute__((ext_vector_type(16))) _Float16 v16h;
typedef __attribute__((ext_vector_type(8)))  float    v8f;

#define GEMM_RELU 1
#define GEMM_ACC  2

__device__ __forceinline__ void atomicMaxF(float* addr, float val) {
    int* ai = (int*)addr;
    int old = __float_as_int(*addr);
    while (val > __int_as_float(old)) {
        int assumed = old;
        old = atomicCAS(ai, assumed, __float_as_int(val));
        if (old == assumed) break;
    }
}

// ---------------------------------------------------------------------------
// WMMA GEMM: D[M,N] = (ACC? D:0) + A[M,K] @ W[K,N] + bias[N], optional ReLU.
// Block = 256 threads (8 waves): 128 rows x 64 cols per block.
// A tile DMA'd global->LDS with CDNA5 async-LDS loads (ASYNCcnt), kept fp32,
// converted to f16 at fragment build.  W panel staged coalesced as f16,
// transposed in LDS so B-fragment K-pairs are contiguous 32-bit reads.
// 8x v_wmma_f32_16x16x32_f16 per 64-wide K chunk.
// ---------------------------------------------------------------------------
__global__ __launch_bounds__(256)
void gemm_wmma(const float* __restrict__ A, const float* __restrict__ W,
               const float* __restrict__ bias, float* __restrict__ D,
               int M, int K, int N, int flags)
{
    __shared__ alignas(16) float    a32[128][68];        // fp32 A tile (padded)
    __shared__ alignas(16) _Float16 b_sh[64][64 + 8];    // b_sh[n][k] f16 (transposed)
    const int tid  = threadIdx.x;
    const int wave = tid >> 5;
    const int lane = tid & 31;
    const int m    = lane & 15;      // row-in-tile (A) / col-in-tile (B,C)
    const int hf   = lane >> 4;      // lane half
    const int rowBase = blockIdx.x * 128;
    const int colBase = blockIdx.y * 64;

    v8f acc[4] = {};

    for (int k0 = 0; k0 < K; k0 += 64) {
        // --- A tile: async DMA global -> LDS (no VGPR round-trip) ---
        for (int i = tid; i < 128 * 64; i += 256) {
            int r = i >> 6, c = i & 63;
            int gr = rowBase + r;
            if (gr < M) {
                unsigned ldsOff = (unsigned)(size_t)&a32[r][c];
                unsigned long long gp =
                    (unsigned long long)(A + (size_t)gr * K + (k0 + c));
                asm volatile("global_load_async_to_lds_b32 %0, %1, off"
                             :: "v"(ldsOff), "v"(gp) : "memory");
            } else {
                a32[r][c] = 0.0f;
            }
        }
        // --- W panel: coalesced global read (consecutive lanes -> consecutive n),
        //     f16 convert, transposed LDS write ---
        for (int i = tid; i < 64 * 64; i += 256) {
            int n = i & 63, c = i >> 6;
            b_sh[n][c] = (_Float16)W[(size_t)(k0 + c) * N + (colBase + n)];
        }
        asm volatile("s_wait_asynccnt 0" ::: "memory");
        __syncthreads();

        #pragma unroll
        for (int kk = 0; kk < 64; kk += 32) {
            // A fragment: 16x32 f16, CDNA5 wave32 layout; cvt fp32->f16 here
            union { v16h h; _Float16 e[16]; } af;
            #pragma unroll
            for (int vv = 0; vv < 8; ++vv) {
                int grp = vv >> 2, w = vv & 3;
                int kl = kk + 16 * grp + 8 * hf + 2 * w;
                const float* p = &a32[(wave << 4) + m][kl];
                af.e[2 * vv]     = (_Float16)p[0];
                af.e[2 * vv + 1] = (_Float16)p[1];
            }
            #pragma unroll
            for (int t = 0; t < 4; ++t) {
                // B fragment: 32x16 f16; K-pairs contiguous because b_sh is [n][k]
                union { v16h h; unsigned u[8]; } bf;
                #pragma unroll
                for (int vv = 0; vv < 8; ++vv) {
                    int kl = kk + 16 * (vv >> 2) + 8 * hf + 2 * (vv & 3);
                    bf.u[vv] = *(const unsigned*)&b_sh[t * 16 + m][kl];
                }
                acc[t] = __builtin_amdgcn_wmma_f32_16x16x32_f16(
                    false, af.h, false, bf.h, (short)0, acc[t], false, false);
            }
        }
        __syncthreads();
    }

    // C layout: VGPR r -> row r (lanes 0-15) / r+8 (lanes 16-31), col = lane&15
    #pragma unroll
    for (int t = 0; t < 4; ++t) {
        #pragma unroll
        for (int r = 0; r < 8; ++r) {
            int row = rowBase + (wave << 4) + r + 8 * hf;
            int col = colBase + t * 16 + m;
            if (row < M) {
                float v = acc[t][r] + bias[col];
                if (flags & GEMM_RELU) v = fmaxf(v, 0.0f);
                size_t idx = (size_t)row * N + col;
                if (flags & GEMM_ACC) v += D[idx];
                D[idx] = v;
            }
        }
    }
}

// ---------------------------------------------------------------------------
__global__ void fill_kernel(float* __restrict__ p, float val, int n) {
    int i = blockIdx.x * blockDim.x + threadIdx.x;
    if (i < n) p[i] = val;
}

__global__ void ln_kernel(const float* __restrict__ h, const float* __restrict__ g,
                          const float* __restrict__ b, float* __restrict__ o, int n)
{
    int i = blockIdx.x * blockDim.x + threadIdx.x;
    if (i >= n) return;
    const float* row = h + (size_t)i * 64;
    float mu = 0.f;
    #pragma unroll 8
    for (int j = 0; j < 64; ++j) mu += row[j];
    mu *= (1.0f / 64.0f);
    float var = 0.f;
    #pragma unroll 8
    for (int j = 0; j < 64; ++j) { float d = row[j] - mu; var += d * d; }
    float inv = rsqrtf(var * (1.0f / 64.0f) + 1e-5f);
    float* orow = o + (size_t)i * 64;
    #pragma unroll 8
    for (int j = 0; j < 64; ++j) orow[j] = (row[j] - mu) * inv * g[j] + b[j];
}

__global__ void edge_simdist(const float* __restrict__ h, const int* __restrict__ src,
                             const int* __restrict__ tgt, float* __restrict__ sim,
                             float* __restrict__ dist, float* __restrict__ segmax,
                             float beta, int E_)
{
    int e = blockIdx.x * blockDim.x + threadIdx.x;
    if (e >= E_) return;
    int s = src[e], t = tgt[e];
    const float* hs = h + (size_t)s * 64;
    const float* ht = h + (size_t)t * 64;
    float sm = 0.f, dd = 0.f;
    #pragma unroll 8
    for (int j = 0; j < 64; ++j) {
        float a = hs[j], b = ht[j];
        sm += a * b;
        float df = a - b; dd += df * df;
    }
    float sb = sm * beta;
    sim[e] = sb;
    dist[e] = sqrtf(dd);
    atomicMaxF(&segmax[t], sb);
}

__global__ void edge_expsum(const float* __restrict__ sim, const int* __restrict__ tgt,
                            const float* __restrict__ segmax, float* __restrict__ segsum,
                            float* __restrict__ alpha, int E_)
{
    int e = blockIdx.x * blockDim.x + threadIdx.x;
    if (e >= E_) return;
    int t = tgt[e];
    float ex = __expf(sim[e] - segmax[t]);
    alpha[e] = ex;
    atomicAdd(&segsum[t], ex);
}

__global__ void edge_agg(const float* __restrict__ alpha, const float* __restrict__ dist,
                         const int* __restrict__ tgt, const float* __restrict__ segsum,
                         float* __restrict__ agg, int E_)
{
    int e = blockIdx.x * blockDim.x + threadIdx.x;
    if (e >= E_) return;
    int t = tgt[e];
    atomicAdd(&agg[t], (alpha[e] / segsum[t]) * dist[e]);
}

// Fused: curvature -> MLP(relu, Wc2) -> LN(no-affine) -> @Wcb, plus q·k attention
// scores.  LN@Wcb folded:  bias_h = (sum_j ce_j*Wcb[j,h] - mu*colsum_h) / sigma.
__global__ __launch_bounds__(128)
void edge_scores(const float* __restrict__ q, const float* __restrict__ kx,
                 const int* __restrict__ src, const int* __restrict__ tgt,
                 const float* __restrict__ agg, const float* __restrict__ dist,
                 const float* __restrict__ Wc1, const float* __restrict__ bc1,
                 const float* __restrict__ Wc2, const float* __restrict__ bc2,
                 const float* __restrict__ Wcb, const float* __restrict__ bcb,
                 float* __restrict__ scores, float* __restrict__ emax, int E_)
{
    __shared__ float Wc2s[64 * 64];
    __shared__ float Wcbs[64 * 4];
    for (int i = threadIdx.x; i < 64 * 64; i += blockDim.x) Wc2s[i] = Wc2[i];
    for (int i = threadIdx.x; i < 64 * 4; i += blockDim.x) Wcbs[i] = Wcb[i];
    __syncthreads();
    int e = blockIdx.x * blockDim.x + threadIdx.x;
    if (e >= E_) return;
    int s = src[e], t = tgt[e];
    float d = dist[e];
    float curv = 1.0f - agg[t] / fmaxf(d, 1e-6f);
    float tt[64];
    #pragma unroll
    for (int i = 0; i < 64; ++i) tt[i] = fmaxf(curv * Wc1[i] + bc1[i], 0.0f);
    float m1 = 0.f, m2 = 0.f;
    float sh[4] = {0.f, 0.f, 0.f, 0.f};
    float cb[4] = {0.f, 0.f, 0.f, 0.f};
    for (int j = 0; j < 64; ++j) {
        float ce = bc2[j];
        #pragma unroll
        for (int i = 0; i < 64; ++i) ce += tt[i] * Wc2s[i * 64 + j];
        m1 += ce; m2 += ce * ce;
        #pragma unroll
        for (int hh = 0; hh < 4; ++hh) {
            float w = Wcbs[j * 4 + hh];
            sh[hh] += ce * w;
            cb[hh] += w;
        }
    }
    float mu  = m1 * (1.0f / 64.0f);
    float inv = rsqrtf(m2 * (1.0f / 64.0f) - mu * mu + 1e-5f);
    #pragma unroll
    for (int hh = 0; hh < 4; ++hh) {
        const float* qs = q  + (size_t)s * 64 + hh * 16;
        const float* kt = kx + (size_t)t * 64 + hh * 16;
        float dot = 0.f;
        #pragma unroll
        for (int dd2 = 0; dd2 < 16; ++dd2) dot += qs[dd2] * kt[dd2];
        float sc = dot * 0.25f + (sh[hh] - mu * cb[hh]) * inv + bcb[hh];  // /sqrt(DH=16)
        scores[(size_t)e * 4 + hh] = sc;
        atomicMaxF(&emax[t * 4 + hh], sc);
    }
}

__global__ void edge_probsum(float* __restrict__ scores, const int* __restrict__ tgt,
                             const float* __restrict__ emax, float* __restrict__ esum, int E_)
{
    int e = blockIdx.x * blockDim.x + threadIdx.x;
    if (e >= E_) return;
    int t = tgt[e];
    #pragma unroll
    for (int hh = 0; hh < 4; ++hh) {
        float ex = __expf(scores[(size_t)e * 4 + hh] - emax[t * 4 + hh]);
        scores[(size_t)e * 4 + hh] = ex;
        atomicAdd(&esum[t * 4 + hh], ex);
    }
}

__global__ void edge_messages(const float* __restrict__ scores, const float* __restrict__ esum,
                              const float* __restrict__ v, const int* __restrict__ src,
                              const int* __restrict__ tgt, float* __restrict__ out, int E_)
{
    int e = blockIdx.x * blockDim.x + threadIdx.x;
    if (e >= E_) return;
    int s = src[e], t = tgt[e];
    #pragma unroll
    for (int hh = 0; hh < 4; ++hh) {
        float p = scores[(size_t)e * 4 + hh] / esum[t * 4 + hh];
        const float* vt = v + (size_t)t * 64 + hh * 16;
        float* os = out + (size_t)s * 64 + hh * 16;
        #pragma unroll
        for (int dd = 0; dd < 16; ++dd) atomicAdd(&os[dd], p * vt[dd]);
    }
}

// Block-local LDS reduction (ds_add_f32) -> 64 global atomics per block
__global__ __launch_bounds__(256)
void pool_kernel(const float* __restrict__ h, float* __restrict__ pooled, int total) {
    __shared__ float acc[64];
    if (threadIdx.x < 64) acc[threadIdx.x] = 0.f;
    __syncthreads();
    int i = blockIdx.x * blockDim.x + threadIdx.x;
    if (i < total) atomicAdd(&acc[i & 63], h[i]);
    __syncthreads();
    if (threadIdx.x < 64) atomicAdd(&pooled[threadIdx.x], acc[threadIdx.x]);
}

__global__ void head_kernel(const float* __restrict__ pooled,
                            const float* __restrict__ W1, const float* __restrict__ b1,
                            const float* __restrict__ W2, const float* __restrict__ b2,
                            float* __restrict__ out, float invN)
{
    float acc = b2[0];
    for (int j = 0; j < 32; ++j) {
        float t = b1[j];
        for (int i = 0; i < 64; ++i) t += pooled[i] * invN * W1[i * 32 + j];
        acc += fmaxf(t, 0.f) * W2[j];
    }
    out[0] = acc;
}

// ---------------------------------------------------------------------------
extern "C" void kernel_launch(void* const* d_in, const int* in_sizes, int n_in,
                              void* d_out, int out_size, void* d_ws, size_t ws_size,
                              hipStream_t stream)
{
    (void)n_in; (void)out_size; (void)ws_size;
    const float* x     = (const float*)d_in[0];
    const int*   ei    = (const int*)d_in[1];
    const float* Wnode = (const float*)d_in[3];
    const float* bnode = (const float*)d_in[4];
    const float* Wc1   = (const float*)d_in[5];
    const float* bc1   = (const float*)d_in[6];
    const float* Wc2   = (const float*)d_in[7];
    const float* bc2   = (const float*)d_in[8];
    const float* Wq = (const float*)d_in[9];  const float* bq = (const float*)d_in[10];
    const float* Wk = (const float*)d_in[11]; const float* bk = (const float*)d_in[12];
    const float* Wv = (const float*)d_in[13]; const float* bv = (const float*)d_in[14];
    const float* Wo = (const float*)d_in[15]; const float* bo = (const float*)d_in[16];
    const float* Wcb = (const float*)d_in[17]; const float* bcb = (const float*)d_in[18];
    const float* Wf1 = (const float*)d_in[19]; const float* bf1 = (const float*)d_in[20];
    const float* Wf2 = (const float*)d_in[21]; const float* bf2 = (const float*)d_in[22];
    const float* g1 = (const float*)d_in[23]; const float* be1 = (const float*)d_in[24];
    const float* g2 = (const float*)d_in[25]; const float* be2 = (const float*)d_in[26];
    const float* Wout1 = (const float*)d_in[27]; const float* bout1 = (const float*)d_in[28];
    const float* Wout2 = (const float*)d_in[29]; const float* bout2 = (const float*)d_in[30];

    const int Nn = in_sizes[0] / 128;
    const int Ee = in_sizes[1] / 2;
    const int* src = ei;
    const int* tgt = ei + Ee;

    float* Wsp = (float*)d_ws;
    size_t o = 0;
    float* h    = Wsp + o; o += (size_t)Nn * 64;
    float* xn   = Wsp + o; o += (size_t)Nn * 64;
    float* qb   = Wsp + o; o += (size_t)Nn * 64;
    float* kb   = Wsp + o; o += (size_t)Nn * 64;
    float* vb   = Wsp + o; o += (size_t)Nn * 64;
    float* outb = Wsp + o; o += (size_t)Nn * 64;
    float* ffn  = Wsp + o; o += (size_t)Nn * 256;
    float* sim  = Wsp + o; o += (size_t)Ee;
    float* dist = Wsp + o; o += (size_t)Ee;
    float* alph = Wsp + o; o += (size_t)Ee;
    float* scor = Wsp + o; o += (size_t)Ee * 4;
    float* smax = Wsp + o; o += (size_t)Nn;
    float* ssum = Wsp + o; o += (size_t)Nn;
    float* agg  = Wsp + o; o += (size_t)Nn;
    float* emax = Wsp + o; o += (size_t)Nn * 4;
    float* esum = Wsp + o; o += (size_t)Nn * 4;
    float* pool = Wsp + o; o += 64;

    auto fill = [&](float* p, float v_, size_t n) {
        fill_kernel<<<dim3((unsigned)((n + 255) / 256)), dim3(256), 0, stream>>>(p, v_, (int)n);
    };
    auto gemm = [&](const float* A, const float* Wm, const float* bias, float* D,
                    int M, int K, int Nc, int flags) {
        dim3 grid((M + 127) / 128, Nc / 64);
        gemm_wmma<<<grid, dim3(256), 0, stream>>>(A, Wm, bias, D, M, K, Nc, flags);
    };
    const int egrid = (Ee + 255) / 256;
    const int ngrid = (Nn + 255) / 256;

    // node encoder
    gemm(x, Wnode, bnode, h, Nn, 128, 64, 0);

    for (int l = 0; l < 4; ++l) {
        float beta = 1.0f * (float)(l + 1) / 4.0f;

        // dynamic curvature (segment softmax by tgt, then Wasserstein agg)
        fill(smax, -1e30f, (size_t)Nn);
        fill(ssum, 0.f, (size_t)Nn);
        fill(agg,  0.f, (size_t)Nn);
        edge_simdist<<<egrid, 256, 0, stream>>>(h, src, tgt, sim, dist, smax, beta, Ee);
        edge_expsum<<<egrid, 256, 0, stream>>>(sim, tgt, smax, ssum, alph, Ee);
        edge_agg<<<egrid, 256, 0, stream>>>(alph, dist, tgt, ssum, agg, Ee);

        // attention
        ln_kernel<<<ngrid, 256, 0, stream>>>(h, g1 + l * 64, be1 + l * 64, xn, Nn);
        gemm(xn, Wq + (size_t)l * 4096, bq + l * 64, qb, Nn, 64, 64, 0);
        gemm(xn, Wk + (size_t)l * 4096, bk + l * 64, kb, Nn, 64, 64, 0);
        gemm(xn, Wv + (size_t)l * 4096, bv + l * 64, vb, Nn, 64, 64, 0);

        fill(emax, -1e30f, (size_t)Nn * 4);
        fill(esum, 0.f, (size_t)Nn * 4);
        edge_scores<<<(Ee + 127) / 128, 128, 0, stream>>>(qb, kb, src, tgt, agg, dist,
            Wc1, bc1, Wc2, bc2, Wcb + (size_t)l * 256, bcb + (size_t)l * 4, scor, emax, Ee);
        edge_probsum<<<egrid, 256, 0, stream>>>(scor, tgt, emax, esum, Ee);
        fill(outb, 0.f, (size_t)Nn * 64);
        edge_messages<<<egrid, 256, 0, stream>>>(scor, esum, vb, src, tgt, outb, Ee);

        // h += out @ Wo + bo
        gemm(outb, Wo + (size_t)l * 4096, bo + l * 64, h, Nn, 64, 64, GEMM_ACC);

        // FFN with residual
        ln_kernel<<<ngrid, 256, 0, stream>>>(h, g2 + l * 64, be2 + l * 64, xn, Nn);
        gemm(xn, Wf1 + (size_t)l * 16384, bf1 + l * 256, ffn, Nn, 64, 256, GEMM_RELU);
        gemm(ffn, Wf2 + (size_t)l * 16384, bf2 + l * 64, h, Nn, 256, 64, GEMM_ACC);
    }

    // mean pool + head
    fill(pool, 0.f, 64);
    pool_kernel<<<(Nn * 64 + 255) / 256, 256, 0, stream>>>(h, pool, Nn * 64);
    head_kernel<<<1, 1, 0, stream>>>(pool, Wout1, bout1, Wout2, bout2,
                                     (float*)d_out, 1.0f / (float)Nn);
}
